// MultiScaleDeformableAttention_48163763258126
// MI455X (gfx1250) — matrix operand order repaired and checked by reading, not compiled, not run on hardware
//
#include <hip/hip_runtime.h>
#include <hip/hip_bf16.h>
#include <hip/hip_fp16.h>

// ---- Problem constants (from reference) ----
#define DMODEL   256
#define NHEADS   8
#define NLEVELS  4
#define NPOINTS  4
#define NBATCH   2
#define NQUERIES 11253
#define SEQTOT   11253      // 92*92 + 46*46 + 23*23 + 12*12
#define DH       32         // DMODEL / NHEADS
#define KDIM     256
#define LDSPITCH 40         // 32 halves + 8 pad -> 80B row stride, conflict-free frags

typedef __attribute__((ext_vector_type(16))) _Float16 v16h;
typedef __attribute__((ext_vector_type(8)))  _Float16 v8h;
typedef __attribute__((ext_vector_type(8)))  float    v8f;

union V16H { v16h v; _Float16 e[16]; };
union V8F  { v8f  v; float    e[8];  };

// Builtin signature (from probe diagnostic): vV4i*1 V4i*3 Ii Ii
typedef int v4i_gcc __attribute__((vector_size(16)));
typedef __attribute__((address_space(1))) v4i_gcc* as1_v4i_p;
typedef __attribute__((address_space(3))) v4i_gcc* as3_v4i_p;

// Async global -> LDS copy of 16B (no VGPR transit, tracked by ASYNCcnt).
__device__ __forceinline__ void async_cp16(const _Float16* g, _Float16* l) {
#if __has_builtin(__builtin_amdgcn_global_load_async_to_lds_b128)
  __builtin_amdgcn_global_load_async_to_lds_b128(
      (as1_v4i_p)(uintptr_t)g,
      (as3_v4i_p)(uint32_t)(uintptr_t)l, 0, 0);
#else
  asm volatile("global_load_async_to_lds_b128 %0, %1, off"
               :: "v"((uint32_t)(uintptr_t)l), "v"(g) : "memory");
#endif
}

__device__ __forceinline__ void wait_async0() {
#if __has_builtin(__builtin_amdgcn_s_wait_asynccnt)
  __builtin_amdgcn_s_wait_asynccnt(0);
#else
  asm volatile("s_wait_asynccnt 0x0" ::: "memory");
#endif
}

// ---- One-shot f32 -> f16 conversion (8 elements / thread) ----
__global__ __launch_bounds__(256)
void cvt_f32_f16(const float* __restrict__ src, _Float16* __restrict__ dst, int n8) {
  const int i = blockIdx.x * blockDim.x + threadIdx.x;
  if (i >= n8) return;
  const float4* s = reinterpret_cast<const float4*>(src) + (size_t)i * 2;
  float4 u = s[0], w = s[1];
  v8h d;
  d[0] = (_Float16)u.x; d[1] = (_Float16)u.y; d[2] = (_Float16)u.z; d[3] = (_Float16)u.w;
  d[4] = (_Float16)w.x; d[5] = (_Float16)w.y; d[6] = (_Float16)w.z; d[7] = (_Float16)w.w;
  reinterpret_cast<v8h*>(dst)[i] = d;
}

// ---- Async double-buffered LDS-staged WMMA GEMM ----
// C[R x Ncols] = A[R x 256] * W[Ncols x 256]^T + bias (A, W f16; C f32).
// Block = 256 threads (8 waves) -> 128x64 output tile. Per K-step of 32:
// async-stage A(128x32)+B(64x32) into the ping-pong buffer for step k+1
// while computing step k (4 v_wmma per step per wave, 32 total).
__global__ __launch_bounds__(256)
void gemm_wmma_f16(const _Float16* __restrict__ A, const _Float16* __restrict__ W,
                   const float* __restrict__ bias, float* __restrict__ C,
                   int R, int Ncols) {
  __shared__ _Float16 lsA[2][128 * LDSPITCH];
  __shared__ _Float16 lsB[2][64 * LDSPITCH];

  const int tid  = threadIdx.x;
  const int lane = tid & 31;
  const int wv   = tid >> 5;                 // 0..7 -> 16-row slice of the strip
  const int nstrips = Ncols >> 6;            // 64-col strips (4 or 2)
  const int sm = blockIdx.x / nstrips;
  const int sn = blockIdx.x - sm * nstrips;
  const int row0 = sm << 7;                  // 128-row strip base
  const int n0   = sn << 6;                  // 64-col strip base
  const int half = lane >> 4, lt = lane & 15;

  // Per-thread staging coordinates (fixed across K-steps).
  // A: 128 rows x 32 halves = 512 x 16B chunks, 2 per thread.
  const int ar0 = tid >> 2;
  const int ar1 = (256 + tid) >> 2;
  const int c8  = (tid & 3) << 3;
  int grow0 = row0 + ar0; if (grow0 >= R) grow0 = R - 1;  // clamp edge strip
  int grow1 = row0 + ar1; if (grow1 >= R) grow1 = R - 1;
  // B: 64 rows x 32 halves = 256 x 16B chunks, 1 per thread.
  const int br = tid >> 2;

  v8f acc[4] = {};

  // Prologue: stage K-step 0 into buffer 0.
  async_cp16(&A[(size_t)grow0 * KDIM + c8],            &lsA[0][ar0 * LDSPITCH + c8]);
  async_cp16(&A[(size_t)grow1 * KDIM + c8],            &lsA[0][ar1 * LDSPITCH + c8]);
  async_cp16(&W[(size_t)(n0 + br) * KDIM + c8],        &lsB[0][br  * LDSPITCH + c8]);

  #pragma unroll
  for (int ks = 0; ks < 8; ++ks) {
    wait_async0();            // my chunks for buffer ks&1 have landed in LDS
    __syncthreads();          // everyone's chunks landed; prev buffer fully read

    if (ks < 7) {             // stage next step into the other buffer
      const int k1 = (ks + 1) * 32, nb = (ks + 1) & 1;
      async_cp16(&A[(size_t)grow0 * KDIM + k1 + c8],     &lsA[nb][ar0 * LDSPITCH + c8]);
      async_cp16(&A[(size_t)grow1 * KDIM + k1 + c8],     &lsA[nb][ar1 * LDSPITCH + c8]);
      async_cp16(&W[(size_t)(n0 + br) * KDIM + k1 + c8], &lsB[nb][br  * LDSPITCH + c8]);
    }

    const int buf = ks & 1;
    // A fragment (16-bit A 16x32): K = 8*half+[0..7] and +16..23 of this step.
    V16H a;
    const int arow = wv * 16 + lt;
    *(v8h*)&a.e[0] = *(const v8h*)&lsA[buf][arow * LDSPITCH + 8 * half];
    *(v8h*)&a.e[8] = *(const v8h*)&lsA[buf][arow * LDSPITCH + 8 * half + 16];

    #pragma unroll
    for (int j = 0; j < 4; ++j) {
      // B fragment (16-bit B 32x16): K = 8*half+[0..7] and +16..23, col = j*16 + lt.
      V16H b;
      const int bcol = j * 16 + lt;
      *(v8h*)&b.e[0] = *(const v8h*)&lsB[buf][bcol * LDSPITCH + 8 * half];
      *(v8h*)&b.e[8] = *(const v8h*)&lsB[buf][bcol * LDSPITCH + 8 * half + 16];
      acc[j] = __builtin_amdgcn_wmma_f32_16x16x32_f16(
          false, a.v, false, b.v, (short)0, acc[j], false, false);
    }
  }

  // Store: C/D layout -> VGPR r holds row m0 + r + 8*half, col = lane&15.
  #pragma unroll
  for (int j = 0; j < 4; ++j) {
    const int col = n0 + j * 16 + lt;
    const float bv = bias[col];
    V8F d; d.v = acc[j];
    #pragma unroll
    for (int r = 0; r < 8; ++r) {
      const int row = row0 + wv * 16 + r + 8 * half;
      if (row < R) C[(size_t)row * Ncols + col] = d.e[r] + bv;
    }
  }
}

// ---- Fused softmax + bilinear gather; one wave32 per (n,q,m), lane = channel.
// Coordinates are wave-uniform (no divergence); each corner load is one
// coalesced 128B transaction. Output written directly as f16 GEMM input.
__global__ __launch_bounds__(256)
void msda_sample_kernel(const float* __restrict__ value,    // (N*SEQ, 256) f32
                        const float* __restrict__ off,      // (N*Q, 256) f32
                        const float* __restrict__ attnlog,  // (N*Q, 128) f32 logits
                        const float* __restrict__ refp,     // (N, Q, L, 2) f32
                        _Float16* __restrict__ sampled) {   // (N*Q, 256) f16
  const int lane = threadIdx.x & 31;
  const int w    = blockIdx.x * (blockDim.x >> 5) + (threadIdx.x >> 5);
  const int total = NBATCH * NQUERIES * NHEADS;
  if (w >= total) return;

  const int m  = w & (NHEADS - 1);
  const int nq = w >> 3;
  const int n  = nq / NQUERIES;

  const int Hs[4]     = {92, 46, 23, 12};
  const int Ws_[4]    = {92, 46, 23, 12};
  const int starts[4] = {0, 8464, 10580, 11109};

  float lg[16];
  const float* ap = attnlog + (size_t)nq * (NHEADS * 16) + m * 16;
  float mx = -1e30f;
  #pragma unroll
  for (int i = 0; i < 16; ++i) { lg[i] = ap[i]; mx = fmaxf(mx, lg[i]); }
  float s = 0.f;
  #pragma unroll
  for (int i = 0; i < 16; ++i) { lg[i] = __expf(lg[i] - mx); s += lg[i]; }
  const float inv = 1.f / s;

  const float* op  = off  + (size_t)nq * DMODEL + m * (NLEVELS * NPOINTS * 2);
  const float* rp  = refp + (size_t)nq * (NLEVELS * 2);
  const float* vch = value + (size_t)n * SEQTOT * DMODEL + m * DH + lane;

  float acc = 0.f;
  #pragma unroll
  for (int l = 0; l < NLEVELS; ++l) {
    const int H = Hs[l], Wl = Ws_[l];
    const float rx = rp[l * 2 + 0], ry = rp[l * 2 + 1];
    const float invW = 1.f / (float)Wl, invH = 1.f / (float)H;
    const float* vlev = vch + (size_t)starts[l] * DMODEL;
    #pragma unroll
    for (int p = 0; p < NPOINTS; ++p) {
      const float aw = lg[l * 4 + p] * inv;
      const float ox = op[l * 8 + p * 2 + 0];
      const float oy = op[l * 8 + p * 2 + 1];
      const float x = (rx + ox * invW) * (float)Wl - 0.5f;
      const float y = (ry + oy * invH) * (float)H  - 0.5f;
      const float x0f = floorf(x), y0f = floorf(y);
      const float lx = x - x0f, ly = y - y0f;
      const int x0 = (int)x0f, y0 = (int)y0f;
      #pragma unroll
      for (int cy = 0; cy < 2; ++cy) {
        const int yi = y0 + cy;
        if (yi < 0 || yi >= H) continue;            // wave-uniform
        const float wy = cy ? ly : (1.f - ly);
        #pragma unroll
        for (int cx = 0; cx < 2; ++cx) {
          const int xi = x0 + cx;
          if (xi < 0 || xi >= Wl) continue;         // wave-uniform
          const float wx = cx ? lx : (1.f - lx);
          acc += (aw * wy * wx) * vlev[(size_t)(yi * Wl + xi) * DMODEL];
        }
      }
    }
  }
  sampled[(size_t)nq * DMODEL + m * DH + lane] = (_Float16)acc;
}

extern "C" void kernel_launch(void* const* d_in, const int* in_sizes, int n_in,
                              void* d_out, int out_size, void* d_ws, size_t ws_size,
                              hipStream_t stream) {
  (void)in_sizes; (void)n_in; (void)out_size; (void)ws_size;
  const float* query  = (const float*)d_in[0];
  const float* refp   = (const float*)d_in[1];
  const float* flat   = (const float*)d_in[2];
  const float* W_off  = (const float*)d_in[5];
  const float* b_off  = (const float*)d_in[6];
  const float* W_attn = (const float*)d_in[7];
  const float* b_attn = (const float*)d_in[8];
  const float* W_val  = (const float*)d_in[9];
  const float* b_val  = (const float*)d_in[10];
  const float* W_out  = (const float*)d_in[11];
  const float* b_out  = (const float*)d_in[12];
  float* out = (float*)d_out;

  const int R = NBATCH * NQUERIES;  // 22506 (== N*SEQ as well)

  // ---- Workspace layout ----
  float* ws        = (float*)d_ws;
  float* value     = ws;                                   // R*256 f32
  float* offbuf    = value  + (size_t)R * 256;             // R*256 f32
  float* attnbuf   = offbuf + (size_t)R * 256;             // R*128 f32
  _Float16* flat_h    = (_Float16*)(attnbuf + (size_t)R * 128);
  _Float16* query_h   = flat_h    + (size_t)R * 256;
  _Float16* sampled_h = query_h   + (size_t)R * 256;
  _Float16* Wval_h    = sampled_h + (size_t)R * 256;
  _Float16* Woff_h    = Wval_h  + 256 * 256;
  _Float16* Wattn_h   = Woff_h  + 256 * 256;
  _Float16* Wout_h    = Wattn_h + 128 * 256;               // total ~92.6 MB

  auto cvt = [&](const float* src, _Float16* dst, size_t n) {
    const int n8 = (int)(n / 8);
    hipLaunchKernelGGL(cvt_f32_f16, dim3((n8 + 255) / 256), dim3(256), 0, stream,
                       src, dst, n8);
  };
  cvt(flat,   flat_h,  (size_t)R * 256);
  cvt(query,  query_h, (size_t)R * 256);
  cvt(W_val,  Wval_h,  256 * 256);
  cvt(W_off,  Woff_h,  256 * 256);
  cvt(W_attn, Wattn_h, 128 * 256);
  cvt(W_out,  Wout_h,  256 * 256);

  auto launch_gemm = [&](const _Float16* A, const _Float16* Wm, const float* b,
                         float* C, int rows, int ncols) {
    const int blocks = ((rows + 127) / 128) * (ncols / 64);
    hipLaunchKernelGGL(gemm_wmma_f16, dim3(blocks), dim3(256), 0, stream,
                       A, Wm, b, C, rows, ncols);
  };

  launch_gemm(flat_h,  Wval_h,  b_val,  value,   R, 256);  // value projection
  launch_gemm(query_h, Woff_h,  b_off,  offbuf,  R, 256);  // sampling offsets
  launch_gemm(query_h, Wattn_h, b_attn, attnbuf, R, 128);  // attention logits

  {
    const int totalWaves = R * NHEADS;
    const int blocks = (totalWaves + 7) / 8;
    hipLaunchKernelGGL(msda_sample_kernel, dim3(blocks), dim3(256), 0, stream,
                       value, offbuf, attnbuf, refp, sampled_h);
  }

  launch_gemm(sampled_h, Wout_h, b_out, out, R, 256);      // output projection
}
